// ComplexAttentionLayer_52836687675810
// MI455X (gfx1250) — matrix operand order repaired and checked by reading, not compiled, not run on hardware
//
#include <hip/hip_runtime.h>
#include <stdint.h>

typedef __attribute__((ext_vector_type(16))) _Float16 v16h;
typedef __attribute__((ext_vector_type(8)))  _Float16 v8h;
typedef __attribute__((ext_vector_type(8)))  float    v8f;

#define B_   8
#define L_   1024
#define H_   8
#define DH_  64
#define D_   512          // H_ * DH_
#define SCALE_ 0.125f     // 1/sqrt(64)
#define SLDP 1028         // padded f32 score row stride (bank-conflict pad)
#define ALDP 1032         // padded f16 prob row stride (keeps 16B align, shifts banks)

// ---------------- helpers ----------------

__device__ inline v8f zero8() {
  v8f z;
#pragma unroll
  for (int i = 0; i < 8; ++i) z[i] = 0.0f;
  return z;
}

// build a 16-element f16 fragment from two 8-element (16-byte) runs
__device__ inline v16h load8x2(const _Float16* p0, const _Float16* p1) {
  v8h a = *(const v8h*)p0;
  v8h b = *(const v8h*)p1;
  return __builtin_shufflevector(a, b, 0,1,2,3,4,5,6,7,8,9,10,11,12,13,14,15);
}

// negate f16 vector by flipping sign bits
__device__ inline v16h neg16(v16h x) {
  union { v16h v; uint32_t u[8]; } t;
  t.v = x;
#pragma unroll
  for (int i = 0; i < 8; ++i) t.u[i] ^= 0x80008000u;
  return t.v;
}

__device__ inline v8f wmma_f16(v16h a, v16h b, v8f c) {
  // (neg_a, A, neg_b, B, c_mod, C, reuse_a, reuse_b)
  return __builtin_amdgcn_wmma_f32_16x16x32_f16(false, a, false, b, (short)0, c,
                                                false, false);
}

// ---------------- kernel 1: convert x to f16 (flat) + f16 transposed [b][h][d][s] ----
// grid = (L/64, H, B), block = 256. Reads x exactly once, coalesced both directions.

__global__ void __launch_bounds__(256) prep_kernel(
    const float* __restrict__ xr, const float* __restrict__ xi,
    _Float16* __restrict__ hr, _Float16* __restrict__ hi,
    _Float16* __restrict__ tr, _Float16* __restrict__ ti) {
  __shared__ _Float16 ltr[64][72];   // [d][s_local], padded
  __shared__ _Float16 lti[64][72];

  const int b  = blockIdx.z;
  const int h  = blockIdx.y;
  const int s0 = blockIdx.x * 64;

  const int dl = threadIdx.x & 63;   // head-dim lane (coalesced reads)
  const int sl = threadIdx.x >> 6;   // 0..3
#pragma unroll 4
  for (int it = 0; it < 16; ++it) {
    const int sloc = 4 * it + sl;
    const size_t gi = ((size_t)(b * L_ + s0 + sloc)) * D_ + h * DH_ + dl;
    _Float16 vr = (_Float16)xr[gi];
    _Float16 vi = (_Float16)xi[gi];
    hr[gi] = vr;                     // flat f16 copy (same layout as x)
    hi[gi] = vi;
    ltr[dl][sloc] = vr;
    lti[dl][sloc] = vi;
  }
  __syncthreads();
  const int sw = threadIdx.x & 63;   // key lane (coalesced writes)
  const int dw = threadIdx.x >> 6;   // 0..3
#pragma unroll 4
  for (int it = 0; it < 16; ++it) {
    const int d = 4 * it + dw;
    const size_t go = (((size_t)(b * H_ + h)) * DH_ + d) * L_ + s0 + sw;
    tr[go] = ltr[d][sw];
    ti[go] = lti[d][sw];
  }
}

// ---------------- kernel 2: fused complex attention ----------------
// grid = (L/16, B), block = 256 threads (8 waves), loops over 8 heads.

__global__ void __launch_bounds__(256) complex_attn_kernel(
    const _Float16* __restrict__ xhr, const _Float16* __restrict__ xhi,
    const _Float16* __restrict__ xtr, const _Float16* __restrict__ xti,
    float* __restrict__ out_re, float* __restrict__ out_im,
    float* __restrict__ attn_re, float* __restrict__ attn_im) {
  extern __shared__ char smem[];
  float*    s_re  = (float*)smem;                       // 16 * SLDP f32
  float*    s_im  = s_re + 16 * SLDP;                   // 16 * SLDP f32
  _Float16* a_lre = (_Float16*)(s_im + 16 * SLDP);      // 16 * ALDP f16
  _Float16* a_lim = a_lre + 16 * ALDP;                  // 16 * ALDP f16
  float*    red   = (float*)(a_lim + 16 * ALDP);        // 2 * 256 f32
  float*    ored  = red + 512;                          // 4*2*8*32 f32

  const int tid  = threadIdx.x;
  const int lane = tid & 31;
  const int w    = tid >> 5;
  const int lrow = lane & 15;
  const int lhi  = lane >> 4;

  const int b  = blockIdx.y;
  const int m0 = blockIdx.x * 16;

  const int sr = tid >> 4;   // softmax row 0..15
  const int sc = tid & 15;   // softmax column phase

  for (int h = 0; h < H_; ++h) {
    // ================= Phase A: scores S = Q Kt (complex) via WMMA =================
    v16h qa_re[2], qa_im[2], qa_imn[2];
    {
      const size_t qbase = ((size_t)(b * L_ + m0 + lrow)) * D_ + h * DH_;
#pragma unroll
      for (int kf = 0; kf < 2; ++kf) {
        const size_t off = qbase + kf * 32 + lhi * 8;   // A-frag: two 8-elem runs, +16 apart
        qa_re[kf]  = load8x2(xhr + off, xhr + off + 16);
        qa_im[kf]  = load8x2(xhi + off, xhi + off + 16);
        qa_imn[kf] = neg16(qa_im[kf]);
      }
    }
    const int scol0 = w * 128;
#pragma unroll
    for (int ct = 0; ct < 8; ++ct) {
      const int s0 = scol0 + ct * 16;
      v8f c_re = zero8();
      v8f c_im = zero8();
#pragma unroll
      for (int kf = 0; kf < 2; ++kf) {
        // B-frag (Kt): lane holds key s0+lrow, two 8-elem d runs, +16 apart
        const size_t kbase =
            ((size_t)(b * L_ + s0 + lrow)) * D_ + h * DH_ + kf * 32 + lhi * 8;
        v16h kb_re = load8x2(xhr + kbase, xhr + kbase + 16);
        v16h kb_im = load8x2(xhi + kbase, xhi + kbase + 16);
        c_re = wmma_f16(qa_re[kf],  kb_re, c_re);   // + qre.kre
        c_re = wmma_f16(qa_imn[kf], kb_im, c_re);   // - qim.kim
        c_im = wmma_f16(qa_re[kf],  kb_im, c_im);   // + qre.kim
        c_im = wmma_f16(qa_im[kf],  kb_re, c_im);   // + qim.kre
      }
      const int crow = lhi * 8;
#pragma unroll
      for (int r = 0; r < 8; ++r) {
        s_re[(crow + r) * SLDP + s0 + lrow] = c_re[r];
        s_im[(crow + r) * SLDP + s0 + lrow] = c_im[r];
      }
    }
    __syncthreads();

    // ================= Phase B: f32 softmax + attn mean + f16 prob tile ============
    float mx_re = -3.0e38f, mx_im = -3.0e38f;
#pragma unroll 4
    for (int j = 0; j < 64; ++j) {
      mx_re = fmaxf(mx_re, s_re[sr * SLDP + sc + 16 * j]);
      mx_im = fmaxf(mx_im, s_im[sr * SLDP + sc + 16 * j]);
    }
    red[sr * 16 + sc]       = mx_re;
    red[256 + sr * 16 + sc] = mx_im;
    __syncthreads();
    float rmx_re = -3.0e38f, rmx_im = -3.0e38f;
#pragma unroll
    for (int k = 0; k < 16; ++k) {
      rmx_re = fmaxf(rmx_re, red[sr * 16 + k]);
      rmx_im = fmaxf(rmx_im, red[256 + sr * 16 + k]);
    }
    __syncthreads();  // release red for reuse
    float sum_re = 0.0f, sum_im = 0.0f;
#pragma unroll 4
    for (int j = 0; j < 64; ++j) {
      const int idx = sr * SLDP + sc + 16 * j;
      float e_re = __expf((s_re[idx] - rmx_re) * SCALE_);
      float e_im = __expf((s_im[idx] - rmx_im) * SCALE_);
      s_re[idx] = e_re;
      s_im[idx] = e_im;
      sum_re += e_re;
      sum_im += e_im;
    }
    red[sr * 16 + sc]       = sum_re;
    red[256 + sr * 16 + sc] = sum_im;
    __syncthreads();
    float ts_re = 0.0f, ts_im = 0.0f;
#pragma unroll
    for (int k = 0; k < 16; ++k) {
      ts_re += red[sr * 16 + k];
      ts_im += red[256 + sr * 16 + k];
    }
    const float inv_re = 1.0f / ts_re;
    const float inv_im = 1.0f / ts_im;
    float* arow_re = attn_re + ((size_t)(b * L_ + m0 + sr)) * L_;
    float* arow_im = attn_im + ((size_t)(b * L_ + m0 + sr)) * L_;
#pragma unroll 4
    for (int j = 0; j < 64; ++j) {
      const int col = sc + 16 * j;
      const int idx = sr * SLDP + col;
      float a1 = s_re[idx] * inv_re;
      float a2 = s_im[idx] * inv_im;
      a_lre[sr * ALDP + col] = (_Float16)a1;
      a_lim[sr * ALDP + col] = (_Float16)a2;
      // block-exclusive rows -> plain read-modify-write, no atomics
      if (h == 0) {
        arow_re[col] = 0.125f * a1;
        arow_im[col] = 0.125f * a2;
      } else {
        arow_re[col] += 0.125f * a1;
        arow_im[col] += 0.125f * a2;
      }
    }
    __syncthreads();

    // ================= Phase C: O = A V (complex) via WMMA, split-K ================
    const int nt = w & 3;        // N-tile (16 of 64 head dims)
    const int kh = w >> 2;       // K-half (512 of 1024 keys)
    const int d0 = nt * 16;
    v8f o_re = zero8();
    v8f o_im = zero8();
    // base for transposed V: xT[b][h][d0+lrow][*]
    const size_t vrow = (((size_t)(b * H_ + h)) * DH_ + d0 + lrow) * L_;
    for (int ks = 0; ks < 16; ++ks) {
      const int sbase = kh * 512 + ks * 32;
      const _Float16* pr = a_lre + lrow * ALDP + sbase + lhi * 8;
      const _Float16* pi = a_lim + lrow * ALDP + sbase + lhi * 8;
      v16h af_re  = load8x2(pr, pr + 16);
      v16h af_im  = load8x2(pi, pi + 16);
      v16h af_imn = neg16(af_im);
      // V B-frags: lane holds dim d0+lrow, two 8-key runs +16 apart from transposed copy
      const size_t vb0 = vrow + sbase + lhi * 8;
      v16h vb_re = load8x2(xtr + vb0, xtr + vb0 + 16);
      v16h vb_im = load8x2(xti + vb0, xti + vb0 + 16);
      o_re = wmma_f16(af_re,  vb_re, o_re);   // + are.vre
      o_re = wmma_f16(af_imn, vb_im, o_re);   // - aim.vim
      o_im = wmma_f16(af_re,  vb_im, o_im);   // + are.vim
      o_im = wmma_f16(af_im,  vb_re, o_im);   // + aim.vre
    }
    if (kh == 1) {
#pragma unroll
      for (int r = 0; r < 8; ++r) {
        ored[((nt * 2 + 0) * 8 + r) * 32 + lane] = o_re[r];
        ored[((nt * 2 + 1) * 8 + r) * 32 + lane] = o_im[r];
      }
    }
    __syncthreads();
    if (kh == 0) {
#pragma unroll
      for (int r = 0; r < 8; ++r) {
        float fre = o_re[r] + ored[((nt * 2 + 0) * 8 + r) * 32 + lane];
        float fim = o_im[r] + ored[((nt * 2 + 1) * 8 + r) * 32 + lane];
        const size_t oidx =
            ((size_t)(b * L_ + m0 + r + 8 * lhi)) * D_ + h * DH_ + d0 + lrow;
        out_re[oidx] = fre;
        out_im[oidx] = fim;
      }
    }
    __syncthreads();  // protect LDS reuse for next head
  }
}

// ---------------- launch ----------------

extern "C" void kernel_launch(void* const* d_in, const int* in_sizes, int n_in,
                              void* d_out, int out_size, void* d_ws, size_t ws_size,
                              hipStream_t stream) {
  const float* x_re = (const float*)d_in[0];
  const float* x_im = (const float*)d_in[1];

  const size_t nBLD = (size_t)B_ * L_ * D_;   // 4,194,304
  const size_t nBLL = (size_t)B_ * L_ * L_;   // 8,388,608

  float* out     = (float*)d_out;
  float* out_re  = out;
  float* out_im  = out + nBLD;
  float* attn_re = out + 2 * nBLD;
  float* attn_im = attn_re + nBLL;

  _Float16* xhr = (_Float16*)d_ws;            // flat f16 [b][s][h*64+d]
  _Float16* xhi = xhr + nBLD;
  _Float16* xtr = xhi + nBLD;                 // transposed f16 [b][h][d][s]
  _Float16* xti = xtr + nBLD;                 // total ws: 32 MB

  dim3 pgrid(L_ / 64, H_, B_);
  prep_kernel<<<pgrid, 256, 0, stream>>>(x_re, x_im, xhr, xhi, xtr, xti);

  const size_t shmem = (size_t)(16 * SLDP * 4) * 2   // f32 score tiles
                     + (size_t)(16 * ALDP * 2) * 2   // f16 prob tiles
                     + 512 * 4                       // softmax reduce scratch
                     + 2048 * 4;                     // split-K output scratch
  hipFuncSetAttribute((const void*)complex_attn_kernel,
                      hipFuncAttributeMaxDynamicSharedMemorySize, (int)shmem);
  dim3 grid(L_ / 16, B_);
  complex_attn_kernel<<<grid, 256, shmem, stream>>>(xhr, xhi, xtr, xti,
                                                    out_re, out_im,
                                                    attn_re, attn_im);
}
